// GraphEncoderSAGE_65773129171089
// MI455X (gfx1250) — matrix operand order repaired and checked by reading, not compiled, not run on hardware
//
#include <hip/hip_runtime.h>
#include <hip/hip_bf16.h>

#define N_NODES  50000
#define N_EDGES  800000
#define N_FEAT   128
#define N_LAYERS 3
#define N_GRAPHS 256

typedef __attribute__((ext_vector_type(16))) __bf16 v16bf;
typedef __attribute__((ext_vector_type(8)))  __bf16 v8bf;
typedef __attribute__((ext_vector_type(8)))  float  v8f;

// ---------------------------------------------------------------------------
// Utility kernels
// ---------------------------------------------------------------------------
__global__ void k_fill_zero(float* __restrict__ p, int n) {
    int i = blockIdx.x * 256 + threadIdx.x;
    int stride = gridDim.x * 256;
    for (; i < n; i += stride) p[i] = 0.0f;
}

__global__ void k_f32_to_bf16(const float* __restrict__ in, __bf16* __restrict__ out, int n) {
    int i = blockIdx.x * 256 + threadIdx.x;
    if (i < n) out[i] = (__bf16)in[i];
}

// degree[dst] += 1 per edge
__global__ void k_degree(const int* __restrict__ ei, float* __restrict__ deg) {
    int e = blockIdx.x * 256 + threadIdx.x;
    if (e < N_EDGES) unsafeAtomicAdd(deg + ei[N_EDGES + e], 1.0f);
}

// ---------------------------------------------------------------------------
// Edge scatter: agg[dst,:] += h[src,:]  (one wave32 per edge, 4 floats/lane)
// ---------------------------------------------------------------------------
__global__ void k_edge_scatter(const int* __restrict__ ei,
                               const float* __restrict__ h,
                               float* __restrict__ agg) {
    int e = blockIdx.x * 8 + (threadIdx.x >> 5);
    if (e >= N_EDGES) return;
    int lane = threadIdx.x & 31;
    int src = ei[e];
    int dst = ei[N_EDGES + e];
    const float4 v = ((const float4*)(h + (size_t)src * N_FEAT))[lane];
    float* a = agg + (size_t)dst * N_FEAT + lane * 4;
    unsafeAtomicAdd(a + 0, v.x);
    unsafeAtomicAdd(a + 1, v.y);
    unsafeAtomicAdd(a + 2, v.z);
    unsafeAtomicAdd(a + 3, v.w);
}

// ---------------------------------------------------------------------------
// Fused SAGE layer GEMM:
//   hout[m,:] = relu?( (agg[m,:]*inv_deg[m]) @ Wl^T + h[m,:] @ Wr^T + b )
// 16-row slab per block; 8 waves; wave w -> cols [16w,16w+16).
// Tiles converted to bf16 ONCE during staging; fragments are pure b128 loads.
// v_wmma_f32_16x16x32_bf16, K=128 in 4 steps x 2 matmuls = 8 WMMA / wave.
// ---------------------------------------------------------------------------
#define LDSB_STRIDE 136  // bf16 row stride: 272B -> rows offset by 4 banks, b128 conflict-free

__global__ void __launch_bounds__(256)
k_sage_gemm(const float* __restrict__ hin,
            const float* __restrict__ agg,
            const float* __restrict__ deg,
            const __bf16* __restrict__ Wl,   // [128,128] bf16 row-major
            const __bf16* __restrict__ Wr,   // [128,128] bf16 row-major
            const float* __restrict__ bias,  // [128]
            float* __restrict__ hout,
            int relu) {
    __shared__ __bf16 AsB[16 * LDSB_STRIDE];
    __shared__ __bf16 HsB[16 * LDSB_STRIDE];

    const int tid = threadIdx.x;
    const int m0  = blockIdx.x * 16;

    // ---- stage 16x128 tiles into LDS as bf16 (agg scaled by 1/max(deg,1)) ----
    {
        const float4* av = (const float4*)(agg + (size_t)m0 * N_FEAT);
        const float4* hv = (const float4*)(hin + (size_t)m0 * N_FEAT);
        int i8  = tid * 8;            // 8 consecutive floats per thread (2048 total)
        int row = i8 >> 7;
        int col = i8 & 127;
        float inv = 1.0f / fmaxf(deg[m0 + row], 1.0f);
        float4 a0 = av[i8 >> 2], a1 = av[(i8 >> 2) + 1];
        float4 h0 = hv[i8 >> 2], h1 = hv[(i8 >> 2) + 1];
        float fa[8] = {a0.x, a0.y, a0.z, a0.w, a1.x, a1.y, a1.z, a1.w};
        float fh[8] = {h0.x, h0.y, h0.z, h0.w, h1.x, h1.y, h1.z, h1.w};
        v8bf pa, ph;
#pragma unroll
        for (int j = 0; j < 8; ++j) {
            pa[j] = (__bf16)(fa[j] * inv);
            ph[j] = (__bf16)fh[j];
        }
        *(v8bf*)(AsB + row * LDSB_STRIDE + col) = pa;
        *(v8bf*)(HsB + row * LDSB_STRIDE + col) = ph;
    }
    __syncthreads();

    const int wave = tid >> 5;
    const int lane = tid & 31;
    const int m    = lane & 15;   // A-fragment row == C column index
    const int half = lane >> 4;   // K sub-block selector
    const int n0   = wave * 16;
    const int n    = lane & 15;

    // accumulator initialised with bias (broadcast over the 8 M-rows per lane)
    v8f c;
    {
        float b = bias[n0 + n];
#pragma unroll
        for (int r = 0; r < 8; ++r) c[r] = b;
    }

    const __bf16* arow  = AsB + m * LDSB_STRIDE;
    const __bf16* hrow  = HsB + m * LDSB_STRIDE;
    const __bf16* wlrow = Wl + (size_t)(n0 + n) * N_FEAT;
    const __bf16* wrrow = Wr + (size_t)(n0 + n) * N_FEAT;

#pragma unroll
    for (int kb = 0; kb < N_FEAT; kb += 32) {
        // A fragment (16-bit 16x32 layout): lane holds contiguous bf16 runs
        //   k = kb + half*16 + g*8 + [0..8), g = 0,1  -> two b128 LDS loads
        v8bf aa0 = *(const v8bf*)(arow + kb + half * 16);
        v8bf aa1 = *(const v8bf*)(arow + kb + half * 16 + 8);
        v8bf ah0 = *(const v8bf*)(hrow + kb + half * 16);
        v8bf ah1 = *(const v8bf*)(hrow + kb + half * 16 + 8);
        v16bf aA = __builtin_shufflevector(aa0, aa1, 0, 1, 2, 3, 4, 5, 6, 7,
                                           8, 9, 10, 11, 12, 13, 14, 15);
        v16bf aH = __builtin_shufflevector(ah0, ah1, 0, 1, 2, 3, 4, 5, 6, 7,
                                           8, 9, 10, 11, 12, 13, 14, 15);
        // B fragment (16-bit 32x16 layout): B[k,n]=W[n,k]; lane reads
        //   16 consecutive bf16 of row n starting at kb + half*16
        v16bf bL = *(const v16bf*)(wlrow + kb + half * 16);
        v16bf bR = *(const v16bf*)(wrrow + kb + half * 16);

        c = __builtin_amdgcn_wmma_f32_16x16x32_bf16(false, aA, false, bL,
                                                    (short)0, c, false, false);
        c = __builtin_amdgcn_wmma_f32_16x16x32_bf16(false, aH, false, bR,
                                                    (short)0, c, false, false);
    }

    // C/D layout: vgpr r -> M = r + 8*half, column = n0 + n
#pragma unroll
    for (int r = 0; r < 8; ++r) {
        float v = c[r];
        if (relu) v = fmaxf(v, 0.0f);
        hout[(size_t)(m0 + half * 8 + r) * N_FEAT + n0 + n] = v;
    }
}

// ---------------------------------------------------------------------------
// Global mean pool: scatter sums + counts (one wave per node)
// ---------------------------------------------------------------------------
__global__ void k_pool_scatter(const int* __restrict__ batch,
                               const float* __restrict__ h,
                               float* __restrict__ pool,
                               float* __restrict__ counts) {
    int nidx = blockIdx.x * 8 + (threadIdx.x >> 5);
    if (nidx >= N_NODES) return;
    int lane = threadIdx.x & 31;
    int g = batch[nidx];
    const float4 v = ((const float4*)(h + (size_t)nidx * N_FEAT))[lane];
    float* p = pool + (size_t)g * N_FEAT + lane * 4;
    unsafeAtomicAdd(p + 0, v.x);
    unsafeAtomicAdd(p + 1, v.y);
    unsafeAtomicAdd(p + 2, v.z);
    unsafeAtomicAdd(p + 3, v.w);
    if (lane == 0) unsafeAtomicAdd(counts + g, 1.0f);
}

__global__ void k_pool_finalize(const float* __restrict__ pool,
                                const float* __restrict__ counts,
                                float* __restrict__ out) {
    int i = blockIdx.x * 256 + threadIdx.x;
    if (i < N_GRAPHS * N_FEAT) {
        int g = i >> 7;
        out[i] = pool[i] / fmaxf(counts[g], 1.0f);
    }
}

// ---------------------------------------------------------------------------
extern "C" void kernel_launch(void* const* d_in, const int* in_sizes, int n_in,
                              void* d_out, int out_size, void* d_ws, size_t ws_size,
                              hipStream_t stream) {
    const float* x     = (const float*)d_in[0];
    const int*   ei    = (const int*)d_in[1];
    const int*   batch = (const int*)d_in[2];
    const float* Wl    = (const float*)d_in[3];
    const float* Wr    = (const float*)d_in[4];
    const float* bl    = (const float*)d_in[5];
    float*       out   = (float*)d_out;

    // workspace carve-up (all offsets stay 32B-aligned)
    char* ws = (char*)d_ws;
    float* h    = (float*)ws;  ws += (size_t)N_NODES * N_FEAT * 4;
    float* agg  = (float*)ws;  ws += (size_t)N_NODES * N_FEAT * 4;
    float* deg  = (float*)ws;  ws += (size_t)N_NODES * 4;           // | contiguous
    float* pool = (float*)ws;  ws += (size_t)N_GRAPHS * N_FEAT * 4; // | zero block
    float* cnts = (float*)ws;  ws += (size_t)N_GRAPHS * 4;          // |
    __bf16* Wlb = (__bf16*)ws; ws += (size_t)N_LAYERS * N_FEAT * N_FEAT * 2;
    __bf16* Wrb = (__bf16*)ws;

    const int zeroStatN = N_NODES + N_GRAPHS * N_FEAT + N_GRAPHS; // deg+pool+cnts
    const int wN = N_LAYERS * N_FEAT * N_FEAT;

    // 1. zero deg/pool/counts (contiguous)
    k_fill_zero<<<512, 256, 0, stream>>>(deg, zeroStatN);
    // 2. weights -> bf16
    k_f32_to_bf16<<<(wN + 255) / 256, 256, 0, stream>>>(Wl, Wlb, wN);
    k_f32_to_bf16<<<(wN + 255) / 256, 256, 0, stream>>>(Wr, Wrb, wN);
    // 3. degrees
    k_degree<<<(N_EDGES + 255) / 256, 256, 0, stream>>>(ei, deg);

    // 4. layers
    for (int l = 0; l < N_LAYERS; ++l) {
        const float* hcur = (l == 0) ? x : h;
        k_fill_zero<<<2048, 256, 0, stream>>>(agg, N_NODES * N_FEAT);
        k_edge_scatter<<<N_EDGES / 8, 256, 0, stream>>>(ei, hcur, agg);
        k_sage_gemm<<<N_NODES / 16, 256, 0, stream>>>(
            hcur, agg, deg,
            Wlb + (size_t)l * N_FEAT * N_FEAT,
            Wrb + (size_t)l * N_FEAT * N_FEAT,
            bl + (size_t)l * N_FEAT,
            h, (l < N_LAYERS - 1) ? 1 : 0);
    }

    // 5. global mean pool
    k_pool_scatter<<<(N_NODES + 7) / 8, 256, 0, stream>>>(batch, h, pool, cnts);
    k_pool_finalize<<<(N_GRAPHS * N_FEAT + 255) / 256, 256, 0, stream>>>(pool, cnts, out);
}
